// SparseConv_82085414961357
// MI455X (gfx1250) — matrix operand order repaired and checked by reading, not compiled, not run on hardware
//
#include <hip/hip_runtime.h>

// SparseConv3D (3x3x3, Cin=16, F=16, SAME, masked) as implicit GEMM on CDNA5 WMMA.
// f16 inputs / f32 accumulate via v_wmma_f32_16x16x32_f16 (wave32).
// Workgroup = 256 threads (8 waves) -> 4x4x16 voxel tile, halo staged in LDS.

typedef __attribute__((ext_vector_type(16))) _Float16 v16h;
typedef __attribute__((ext_vector_type(8)))  _Float16 v8h;
typedef __attribute__((ext_vector_type(8)))  float    v8f;
typedef __attribute__((ext_vector_type(4)))  float    v4f;

#define DGRID 96
#define CIN   16
#define FOUT  16
#define TAPS  27
#define NCHUNK 14          // ceil(27 taps / 2 taps-per-K32-chunk)
#define HXD 6
#define HYD 6
#define HZD 18

__global__ __launch_bounds__(256)
void sparse_conv3d_wmma(const float* __restrict__ feat,
                        const int*   __restrict__ indexv,
                        const float* __restrict__ w,
                        const float* __restrict__ bias,
                        float*       __restrict__ out)
{
    // halo features, f16: [hx][hy][hz][c]
    __shared__ __align__(32) _Float16 sfeat[HXD * HYD * HZD * CIN];
    // transposed weights, f16: [tap][f][c]; tap 27 is zero padding
    __shared__ __align__(32) _Float16 swT[(NCHUNK * 2) * FOUT * CIN];

    const int tid  = threadIdx.x;
    const int lane = tid & 31;
    const int wave = tid >> 5;

    // block -> (batch, x-tile, y-tile, z-tile)
    int id = blockIdx.x;
    const int bi = id / (24 * 24 * 6);  id %= (24 * 24 * 6);
    const int xt = id / (24 * 6);       id %= (24 * 6);
    const int yt = id / 6;
    const int zt = id % 6;
    const int x0 = xt * 4, y0 = yt * 4, z0 = zt * 16;

    // ---- stage weights: w[(tap*16+c)*16+f] -> swT[tap][f][c], f32->f16 ----
    for (int v = tid; v < (NCHUNK * 2) * FOUT * CIN; v += 256) {
        const int t = v >> 8;
        const int f = (v >> 4) & 15;
        const int c = v & 15;
        float val = 0.0f;
        if (t < TAPS) val = w[(t * CIN + c) * FOUT + f];
        swT[v] = (_Float16)val;
    }

    // ---- stage feature halo (6x6x18 voxels x 16ch), f32->f16, OOB=0 ----
    for (int v = tid; v < HXD * HYD * HZD; v += 256) {
        const int hx = v / (HYD * HZD);
        const int hy = (v / HZD) % HYD;
        const int hz = v % HZD;
        const int gx = x0 + hx - 1;
        const int gy = y0 + hy - 1;
        const int gz = z0 + hz - 1;
        v8h lo = {}, hi = {};
        if ((unsigned)gx < DGRID && (unsigned)gy < DGRID && (unsigned)gz < DGRID) {
            const v4f* src4 = (const v4f*)(feat +
                (size_t)(((bi * DGRID + gx) * DGRID + gy) * DGRID + gz) * CIN);
            v4f q0 = src4[0], q1 = src4[1], q2 = src4[2], q3 = src4[3];
            #pragma unroll
            for (int c = 0; c < 4; ++c) {
                lo[c]     = (_Float16)q0[c];
                lo[c + 4] = (_Float16)q1[c];
                hi[c]     = (_Float16)q2[c];
                hi[c + 4] = (_Float16)q3[c];
            }
        }
        _Float16* dst = sfeat + (size_t)v * CIN;
        *(v8h*)(dst)     = lo;   // ds_store_b128
        *(v8h*)(dst + 8) = hi;   // ds_store_b128
    }
    __syncthreads();

    // A-fragment lane roles (16-bit A 16x32, ISA 7.12.2):
    //  lane<16 : row M=lane,    holds K 0..7 & 16..23  -> channels 0..7 of both taps
    //  lane>=16: row M=lane-16, holds K 8..15 & 24..31 -> channels 8..15 of both taps
    const int r  = lane & 15;          // voxel row within the M tile
    const int ch = (lane >> 4) * 8;    // channel half this lane supplies
    const int n  = lane & 15;          // output filter column (B/C/D layouts)
    const float bv = bias[n];

    // each wave computes 2 of the 16 (lx,ly) M-tiles
    #pragma unroll
    for (int ti = 0; ti < 2; ++ti) {
        const int mt = wave * 2 + ti;
        const int lx = mt >> 2;        // 0..3
        const int ly = mt & 3;         // 0..3

        v8f acc = {};
        #pragma unroll
        for (int k = 0; k < NCHUNK; ++k) {
            const int t0  = 2 * k;
            const int t1  = 2 * k + 1;
            const int t1c = (t1 > TAPS - 1) ? (TAPS - 1) : t1; // pad tap: B rows are 0
            const int dx0 = t0 / 9,  dy0 = (t0 / 3) % 3,  dz0 = t0 % 3;
            const int dx1 = t1c / 9, dy1 = (t1c / 3) % 3, dz1 = t1c % 3;

            // A fragment: two contiguous 16B LDS reads (channels ch..ch+7 of each tap)
            const _Float16* a0p = sfeat +
                ((((lx + dx0) * HYD + (ly + dy0)) * HZD + (r + dz0)) * CIN + ch);
            const _Float16* a1p = sfeat +
                ((((lx + dx1) * HYD + (ly + dy1)) * HZD + (r + dz1)) * CIN + ch);
            v8h alo = *(const v8h*)a0p;
            v8h ahi = *(const v8h*)a1p;
            v16h A;
            #pragma unroll
            for (int i = 0; i < 8; ++i) { A[i] = alo[i]; A[8 + i] = ahi[i]; }

            // B fragment: channels ch..ch+7 of tap t0 then of tap t1 (matches A K order)
            const v8h b0 = *(const v8h*)(swT + ((size_t)t0 * FOUT + n) * CIN + ch);
            const v8h b1 = *(const v8h*)(swT + ((size_t)t1 * FOUT + n) * CIN + ch);
            const v16h Bf = __builtin_shufflevector(b0, b1, 0, 1, 2, 3, 4, 5, 6, 7, 8, 9, 10, 11, 12, 13, 14, 15);

            acc = __builtin_amdgcn_wmma_f32_16x16x32_f16(
                false, A, false, Bf, (short)0, acc, false, false);
        }

        // C/D layout: VGPR i -> M = i + (lane<16 ? 0 : 8), N = lane%16
        const int gx = x0 + lx, gy = y0 + ly;
        const size_t rowbase = (size_t)((bi * DGRID + gx) * DGRID + gy) * DGRID;
        const int mb = (lane >> 4) * 8;
        #pragma unroll
        for (int i = 0; i < 8; ++i) {
            const int gz  = z0 + mb + i;
            const int msk = indexv[rowbase + gz];
            const float v = msk ? (acc[i] + bv) : 0.0f;
            out[(rowbase + gz) * FOUT + n] = v;
        }
    }
}

extern "C" void kernel_launch(void* const* d_in, const int* in_sizes, int n_in,
                              void* d_out, int out_size, void* d_ws, size_t ws_size,
                              hipStream_t stream) {
    (void)in_sizes; (void)n_in; (void)out_size; (void)d_ws; (void)ws_size;
    const float* feat   = (const float*)d_in[0];
    const int*   indexv = (const int*)d_in[1];
    const float* w      = (const float*)d_in[2];
    const float* bias   = (const float*)d_in[3];
    float* out = (float*)d_out;

    const int nblocks = 4 * 24 * 24 * 6;  // B * (96/4) * (96/4) * (96/16)
    sparse_conv3d_wmma<<<nblocks, 256, 0, stream>>>(feat, indexv, w, bias, out);
}
